// MultiHeadAttention_65773129171660
// MI455X (gfx1250) — matrix operand order repaired and checked by reading, not compiled, not run on hardware
//
#include <hip/hip_runtime.h>

// MHA forward for MI455X (gfx1250, wave32).
// Dims fixed by the reference: B=4, N=2048, DIM=1024, H=16, D=64.
#define Bz   4
#define Nz   2048
#define DIMz 1024
#define Hz   16
#define Dz   64
#define ROWS (Bz * Nz)        // 8192 token rows
#define BHz  (Bz * Hz)        // 64

typedef _Float16 v8h  __attribute__((ext_vector_type(8)));
typedef _Float16 v16h __attribute__((ext_vector_type(16)));
typedef float    v8f  __attribute__((ext_vector_type(8)));
typedef unsigned int uint32x4 __attribute__((ext_vector_type(4)));
typedef int          int32x8  __attribute__((ext_vector_type(8)));
typedef int          int32x4  __attribute__((ext_vector_type(4)));

__device__ __forceinline__ v16h cat8(v8h lo, v8h hi) {
  return __builtin_shufflevector(lo, hi, 0,1,2,3,4,5,6,7,8,9,10,11,12,13,14,15);
}

// A-fragment, 16x32 f16 (M x K), source row-major with leading dim ld.
// lane m = lane&15 holds row m; halves = K kb..kb+7 and kb+16..kb+23, kb = 8*(lane>=16).
__device__ __forceinline__ v16h load_frag_a(const _Float16* __restrict__ base, int ld) {
  const int lane = threadIdx.x & 31;
  const int m  = lane & 15;
  const int kb = (lane >> 4) << 3;
  const _Float16* p = base + m * ld + kb;
  v8h lo = *(const v8h*)(p);
  v8h hi = *(const v8h*)(p + 16);
  return cat8(lo, hi);
}

// B-fragment, 32x16 f16 (K x N), source is B-transposed row-major: baseT[n][k].
// lane n = lane&15 holds column n; halves = K kb..kb+7 and kb+16..kb+23, kb = 8*(lane>=16).
__device__ __forceinline__ v16h load_frag_bt(const _Float16* baseT, int ld) {
  const int lane = threadIdx.x & 31;
  const int n  = lane & 15;
  const int kb = (lane >> 4) << 3;
  const _Float16* p = baseT + n * ld + kb;
  v8h lo = *(const v8h*)(p);
  v8h hi = *(const v8h*)(p + 16);
  return cat8(lo, hi);
}

__device__ __forceinline__ v8f wmma16(v16h a, v16h b, v8f c) {
  return __builtin_amdgcn_wmma_f32_16x16x32_f16(false, a, false, b, (short)0, c, false, false);
}

// -------- Tensor Data Mover: 2-D tile load, D# per CDNA5 ISA ch.8 (6-arg clang-23 form) ---
// group0: w0 = count(1);  w1 = lds byte addr;  w2 = global_addr lo;  w3 = ga hi | type=2<<30
// group1: w0 = data_size=2B(<<16) | pad_en(<<20) | pad_interval(<<22) | pad_amount(<<25)
//         w1 = tensor_dim0 << 16 ; w2 = tensor_dim1 << 16 ; w3 = tile_dim0 << 16
//         w4 = tile_dim1 ; w5 = tensor_dim0_stride (lo32) ; w6 = w7 = 0
__device__ __forceinline__ void tdm_load_2d(unsigned lds_addr, const void* gptr,
                                            int w0, int w1, int w2, int w3, int w4, int w5) {
  unsigned long long ga = (unsigned long long)(uintptr_t)gptr;
  uint32x4 g0;
  g0.x = 1u;                                            // count=1, user descriptor
  g0.y = lds_addr;
  g0.z = (unsigned)(ga & 0xFFFFFFFFu);
  g0.w = (unsigned)((ga >> 32) & 0x01FFFFFFu) | 0x80000000u;   // type=2 ("image")
  int32x8 g1 = { w0, w1, w2, w3, w4, w5, 0, 0 };
  int32x4 z4 = { 0, 0, 0, 0 };
  int32x8 z8 = { 0, 0, 0, 0, 0, 0, 0, 0 };
  __builtin_amdgcn_tensor_load_to_lds(g0, g1, z4, z4, z8, 0);
}

__device__ __forceinline__ unsigned lds_byte_addr(const void* p) {
  return (unsigned)(uintptr_t)p;   // low 32 bits of LDS-aperture address = LDS offset
}

// ---------------- prep: f32 -> f16 cast, and f32 -> f16 transpose ----------------
__global__ void __launch_bounds__(256) cvt_f16_kernel(const float* __restrict__ src,
                                                      _Float16* __restrict__ dst, int n) {
  int i = blockIdx.x * 256 + threadIdx.x;
  if (i < n) dst[i] = (_Float16)src[i];
}

__global__ void __launch_bounds__(256) transpose_f16_kernel(const float* __restrict__ src,
                                                            _Float16* __restrict__ dst,
                                                            int rows, int cols) {
  int i = blockIdx.x * 256 + threadIdx.x;
  if (i < rows * cols) {
    int r = i / cols, c = i - r * cols;
    dst[c * rows + r] = (_Float16)src[i];   // dst[cols][rows]
  }
}

// ---------------- stage 1: QKV GEMM (8192x3072x1024) + RoPE, scatter Q/K/Vt ----------------
// One wave per 32x64 output tile (B panel reused 2x). grid = (3072/64, 8192/32), block = 32.
__global__ void __launch_bounds__(32) qkv_rope_kernel(
    const _Float16* __restrict__ XH,    // [8192][1024]
    const _Float16* __restrict__ WQT,   // [3072][1024]  (= W_qkv^T, f16)
    const float*    __restrict__ fcos,  // [2048][32]
    const float*    __restrict__ fsin,  // [2048][32]
    _Float16* __restrict__ Qb,          // [B*H][N][64]
    _Float16* __restrict__ Kb,          // [B*H][N][64]
    _Float16* __restrict__ VT)          // [B*H][64][N]
{
  const int c0   = blockIdx.x * 64;     // output column base (0..3071)
  const int row0 = blockIdx.y * 32;     // token row base (0..8191)
  const int lane = threadIdx.x;

  v8f acc[2][4] = {};
  const _Float16* arow0 = XH + (size_t)row0 * DIMz;
  const _Float16* arow1 = XH + (size_t)(row0 + 16) * DIMz;
  for (int k = 0; k < DIMz; k += 32) {
    if (k + 32 < DIMz) {
      __builtin_prefetch((const void*)(arow0 + (lane & 15) * DIMz + k + 32), 0, 1);
      __builtin_prefetch((const void*)(arow1 + (lane & 15) * DIMz + k + 32), 0, 1);
    }
    v16h a0 = load_frag_a(arow0 + k, DIMz);
    v16h a1 = load_frag_a(arow1 + k, DIMz);
#pragma unroll
    for (int t = 0; t < 4; ++t) {
      v16h b = load_frag_bt(WQT + (size_t)(c0 + t * 16) * DIMz + k, DIMz);
      acc[0][t] = wmma16(a0, b, acc[0][t]);
      acc[1][t] = wmma16(a1, b, acc[1][t]);
    }
  }

  const int sec     = c0 >> 10;            // 0=q, 1=k, 2=v (uniform per block)
  const int h       = (c0 & 1023) >> 6;    // head (uniform per block)
  const int colLane = lane & 15;
  const int rowOff  = (lane >> 4) << 3;

#pragma unroll
  for (int rr = 0; rr < 2; ++rr) {
#pragma unroll
    for (int t = 0; t < 4; ++t) {
      const int d = t * 16 + colLane;      // 0..63 within head
#pragma unroll
      for (int r = 0; r < 8; ++r) {
        const int gr   = row0 + rr * 16 + r + rowOff;
        const int bidx = gr >> 11;
        const int n    = gr & (Nz - 1);
        float v = acc[rr][t][r];
        float o;
        if (sec < 2) {                     // rope(q), rope(k)
          float other = __shfl_xor(v, 1, 32);
          const int fi = n * (Dz / 2) + (d >> 1);
          float c = fcos[fi], s = fsin[fi];
          o = (d & 1) ? (other * s + v * c) : (v * c - other * s);
        } else {
          o = v;
        }
        const int bh = bidx * Hz + h;
        if (sec == 0)      Qb[((size_t)bh * Nz + n) * Dz + d] = (_Float16)o;
        else if (sec == 1) Kb[((size_t)bh * Nz + n) * Dz + d] = (_Float16)o;
        else               VT[((size_t)bh * Dz + d) * Nz + n] = (_Float16)o;
      }
    }
  }
}

// ---------------- stage 2: flash attention ----------------
// Workgroup = 4 waves; each wave owns a 16-query tile; WG covers 64 queries of one (b,h).
// K/V tiles for each 32-key chunk staged once per WG in LDS by the TDM, double-buffered.
// grid = (N/64, B*H), block = 128.
#define CHK 32          // keys per chunk
#define KLD 72          // K tile LDS stride (halves): 64 + 8 pad -> aligned, conflict-free
#define VLD 40          // V/P tile LDS stride (halves): 32 + 8 pad
__global__ void __launch_bounds__(128) attn_kernel(
    const _Float16* __restrict__ Qb,
    const _Float16* __restrict__ Kb,
    const _Float16* __restrict__ VT,
    _Float16* __restrict__ O)            // [8192][1024]
{
  __shared__ __align__(16) _Float16 shK[2][CHK * KLD];   //  9216 B
  __shared__ __align__(16) _Float16 shV[2][Dz * VLD];    // 10240 B
  __shared__ __align__(16) _Float16 shP[4][16 * VLD];    //  5120 B

  const int bh   = blockIdx.y;
  const int wid  = threadIdx.x >> 5;
  const int lane = threadIdx.x & 31;
  const int q0   = blockIdx.x * 64 + wid * 16;

  const _Float16* Qbh = Qb + (size_t)bh * Nz * Dz;
  const _Float16* Kbh = Kb + (size_t)bh * Nz * Dz;
  const _Float16* Vbh = VT + (size_t)bh * Dz * Nz;

  // K-tile D# group1: 2B elems, pad 4 DWORDs every 128 B (per 64-half row):
  //   interval code 4 (2^(4+3)=128B), amount code 3 (4 DWORDs = 16B) -> stride 72 halves.
  const int K_w0 = (1 << 16) | (1 << 20) | (4 << 22) | (3 << 25);
  // V-tile: pad 4 DWORDs every 64 B (per 32-half row) -> stride 40 halves.
  const int V_w0 = (1 << 16) | (1 << 20) | (3 << 22) | (3 << 25);

  // Q tile as two A-fragments; reused for the whole key loop.
  const v16h aq0 = load_frag_a(Qbh + (size_t)q0 * Dz + 0,  Dz);
  const v16h aq1 = load_frag_a(Qbh + (size_t)q0 * Dz + 32, Dz);

  v8f oacc[4] = {};
  float rowmax[8], rowsum[8];
#pragma unroll
  for (int r = 0; r < 8; ++r) { rowmax[r] = -3.0e30f; rowsum[r] = 0.0f; }
  const float scale = 0.125f;            // 1/sqrt(64)

  const int colLane = lane & 15;
  const int rowOff  = (lane >> 4) << 3;

  // Preload chunk 0 (wave 0 issues; TDM ignores EXEC; wave-uniform branch keeps EXEC full).
  if (wid == 0) {
    tdm_load_2d(lds_byte_addr(&shK[0][0]), Kbh,
                K_w0, Dz << 16, Nz << 16, Dz << 16, CHK, Dz);
    tdm_load_2d(lds_byte_addr(&shV[0][0]), Vbh,
                V_w0, Nz << 16, Dz << 16, CHK << 16, Dz, Nz);
    __builtin_amdgcn_s_wait_tensorcnt(0);
  }
  __syncthreads();

  for (int kc = 0; kc < Nz; kc += CHK) {
    const int buf = (kc >> 5) & 1;
    if (wid == 0 && (kc + CHK) < Nz) {           // kick off next chunk into other buffer
      tdm_load_2d(lds_byte_addr(&shK[buf ^ 1][0]), Kbh + (size_t)(kc + CHK) * Dz,
                  K_w0, Dz << 16, Nz << 16, Dz << 16, CHK, Dz);
      tdm_load_2d(lds_byte_addr(&shV[buf ^ 1][0]), Vbh + (kc + CHK),
                  V_w0, Nz << 16, Dz << 16, CHK << 16, Dz, Nz);
    }
    const _Float16* kT = &shK[buf][0];           // [32 keys][64 d] stride KLD
    const _Float16* vT = &shV[buf][0];           // [64 d][32 keys] stride VLD

    // S = Q K^T for this chunk: two 16x16 tiles, K-dim = D via 2 chunks of 32.
    v8f s0 = {}, s1 = {};
    s0 = wmma16(aq0, load_frag_bt(kT + 0 * KLD + 0,  KLD), s0);
    s0 = wmma16(aq1, load_frag_bt(kT + 0 * KLD + 32, KLD), s0);
    s1 = wmma16(aq0, load_frag_bt(kT + 16 * KLD + 0,  KLD), s1);
    s1 = wmma16(aq1, load_frag_bt(kT + 16 * KLD + 32, KLD), s1);

    // Online softmax; row stats per (r, lane-half), reduced across the 16-lane half.
    float p0[8], p1[8];
#pragma unroll
    for (int r = 0; r < 8; ++r) {
      float a0 = s0[r] * scale, a1 = s1[r] * scale;
      float m = fmaxf(a0, a1);
      m = fmaxf(m, __shfl_xor(m, 1, 32));
      m = fmaxf(m, __shfl_xor(m, 2, 32));
      m = fmaxf(m, __shfl_xor(m, 4, 32));
      m = fmaxf(m, __shfl_xor(m, 8, 32));
      float mnew  = fmaxf(rowmax[r], m);
      float alpha = __expf(rowmax[r] - mnew);
      rowmax[r] = mnew;
      p0[r] = __expf(a0 - mnew);
      p1[r] = __expf(a1 - mnew);
      float ps = p0[r] + p1[r];
      ps += __shfl_xor(ps, 1, 32);
      ps += __shfl_xor(ps, 2, 32);
      ps += __shfl_xor(ps, 4, 32);
      ps += __shfl_xor(ps, 8, 32);
      rowsum[r] = rowsum[r] * alpha + ps;
#pragma unroll
      for (int t = 0; t < 4; ++t) oacc[t][r] *= alpha;
    }

    // P (C-layout) -> per-wave LDS tile (same-wave DS ops are in-order; no barrier needed).
    _Float16* pw = &shP[wid][0];
#pragma unroll
    for (int r = 0; r < 8; ++r) {
      pw[(r + rowOff) * VLD + colLane]      = (_Float16)p0[r];
      pw[(r + rowOff) * VLD + colLane + 16] = (_Float16)p1[r];
    }
    const _Float16* pp = pw + (lane & 15) * VLD + ((lane >> 4) << 3);
    v16h pa = cat8(*(const v8h*)(pp), *(const v8h*)(pp + 16));

    // O += P V  (K-dim = 32 keys; B^T rows = V^T d-rows in LDS).
#pragma unroll
    for (int t = 0; t < 4; ++t) {
      oacc[t] = wmma16(pa, load_frag_bt(vT + (t * 16) * VLD, VLD), oacc[t]);
    }

    if (wid == 0) __builtin_amdgcn_s_wait_tensorcnt(0);  // next chunk landed
    __syncthreads();                                     // all waves done with this buffer
  }

  // Normalize and store O[b][q][h*64 + d] as f16.
  const int b = bh >> 4;
  const int h = bh & 15;
#pragma unroll
  for (int t = 0; t < 4; ++t) {
#pragma unroll
    for (int r = 0; r < 8; ++r) {
      const int q  = q0 + r + rowOff;
      const int gr = b * Nz + q;
      float v = oacc[t][r] / rowsum[r];
      O[(size_t)gr * DIMz + h * Dz + t * 16 + colLane] = (_Float16)v;
    }
  }
}

// ---------------- stage 3: output projection (8192x1024x1024) + bias ----------------
// One wave per 32x64 tile. grid = (1024/64, 8192/32), block = 32.
__global__ void __launch_bounds__(32) proj_kernel(
    const _Float16* __restrict__ O,     // [8192][1024]
    const _Float16* __restrict__ WPT,   // [1024][1024] (= W_proj^T, f16)
    const float*    __restrict__ bias,  // [1024]
    float* __restrict__ out)            // [8192][1024] f32
{
  const int c0   = blockIdx.x * 64;
  const int row0 = blockIdx.y * 32;
  const int lane = threadIdx.x;

  v8f acc[2][4] = {};
  const _Float16* arow0 = O + (size_t)row0 * DIMz;
  const _Float16* arow1 = O + (size_t)(row0 + 16) * DIMz;
  for (int k = 0; k < DIMz; k += 32) {
    if (k + 32 < DIMz) {
      __builtin_prefetch((const void*)(arow0 + (lane & 15) * DIMz + k + 32), 0, 1);
      __builtin_prefetch((const void*)(arow1 + (lane & 15) * DIMz + k + 32), 0, 1);
    }
    v16h a0 = load_frag_a(arow0 + k, DIMz);
    v16h a1 = load_frag_a(arow1 + k, DIMz);
#pragma unroll
    for (int t = 0; t < 4; ++t) {
      v16h b = load_frag_bt(WPT + (size_t)(c0 + t * 16) * DIMz + k, DIMz);
      acc[0][t] = wmma16(a0, b, acc[0][t]);
      acc[1][t] = wmma16(a1, b, acc[1][t]);
    }
  }

  const int colLane = lane & 15;
  const int rowOff  = (lane >> 4) << 3;
#pragma unroll
  for (int rr = 0; rr < 2; ++rr) {
#pragma unroll
    for (int t = 0; t < 4; ++t) {
      const int c  = c0 + t * 16 + colLane;
      const float bv = bias[c];
#pragma unroll
      for (int r = 0; r < 8; ++r) {
        out[(size_t)(row0 + rr * 16 + r + rowOff) * DIMz + c] = acc[rr][t][r] + bv;
      }
    }
  }
}

// ---------------- host side ----------------
extern "C" void kernel_launch(void* const* d_in, const int* in_sizes, int n_in,
                              void* d_out, int out_size, void* d_ws, size_t ws_size,
                              hipStream_t stream) {
  (void)in_sizes; (void)n_in; (void)out_size; (void)ws_size;

  const float* x     = (const float*)d_in[0];   // [4,2048,1024]
  const float* fcos  = (const float*)d_in[1];   // [2048,32]
  const float* fsin  = (const float*)d_in[2];   // [2048,32]
  const float* Wqkv  = (const float*)d_in[3];   // [1024,3072]
  const float* Wproj = (const float*)d_in[4];   // [1024,1024]
  const float* bproj = (const float*)d_in[5];   // [1024]
  float* out = (float*)d_out;

  char* ws = (char*)d_ws;
  size_t off = 0;
  _Float16* XH  = (_Float16*)(ws + off); off += (size_t)ROWS * DIMz * 2;        // 16 MB
  _Float16* WQT = (_Float16*)(ws + off); off += (size_t)3 * DIMz * DIMz * 2;    //  6 MB
  _Float16* WPT = (_Float16*)(ws + off); off += (size_t)DIMz * DIMz * 2;        //  2 MB
  _Float16* Qb  = (_Float16*)(ws + off); off += (size_t)BHz * Nz * Dz * 2;      // 16 MB
  _Float16* Kb  = (_Float16*)(ws + off); off += (size_t)BHz * Nz * Dz * 2;      // 16 MB
  _Float16* VT  = (_Float16*)(ws + off); off += (size_t)BHz * Dz * Nz * 2;      // 16 MB
  _Float16* Ob  = (_Float16*)(ws + off); off += (size_t)ROWS * DIMz * 2;        // 16 MB

  const int nx = ROWS * DIMz;
  cvt_f16_kernel<<<(nx + 255) / 256, 256, 0, stream>>>(x, XH, nx);

  const int nwq = DIMz * 3 * DIMz;
  transpose_f16_kernel<<<(nwq + 255) / 256, 256, 0, stream>>>(Wqkv, WQT, DIMz, 3 * DIMz);

  const int nwp = DIMz * DIMz;
  transpose_f16_kernel<<<(nwp + 255) / 256, 256, 0, stream>>>(Wproj, WPT, DIMz, DIMz);

  qkv_rope_kernel<<<dim3(3 * DIMz / 64, ROWS / 32), 32, 0, stream>>>(
      XH, WQT, fcos, fsin, Qb, Kb, VT);

  attn_kernel<<<dim3(Nz / 64, BHz), 128, 0, stream>>>(Qb, Kb, VT, Ob);

  proj_kernel<<<dim3(DIMz / 64, ROWS / 32), 32, 0, stream>>>(Ob, WPT, bproj, out);
}
